// HyperbolicVQ_24739011625044
// MI455X (gfx1250) — matrix operand order repaired and checked by reading, not compiled, not run on hardware
//
#include <hip/hip_runtime.h>
#include <hip/hip_bf16.h>

typedef __attribute__((ext_vector_type(16))) _Float16 v16h;
typedef __attribute__((ext_vector_type(8)))  float    v8f;
typedef __attribute__((ext_vector_type(4)))  unsigned int u32x4;

#define NN 32768
#define DD 512
#define KK 8192
#define DECAYF 0.99f
#define COMMITF 0.25f
#define EPSF 1e-5f

// ---- workspace layout (offsets in floats) ----
#define O_FLATF32 0ull                 // 16,777,216  normalized inputs f32
#define O_EMBF32  16777216ull          //  4,194,304  normalized codebook f32
#define O_DW      20971520ull          //  4,194,304  dw scatter accumulator
#define O_COUNTS  25165824ull          //      8,192
#define O_LOSSA   25174016ull          //          1
#define O_NA      25174017ull          //          1
#define O_PRE     25174018ull          //      8,192  new_cs pre-normalization
#define O_IDX     25182212ull          //     32,768  (ints)
#define O_F16FLAT 25215232ull          //  8,388,608 floats = 16,777,216 halves
#define O_F16EMB  33603840ull          //  2,097,152 floats =  4,194,304 halves

// ---- output layout (offsets in floats, reference return order) ----
#define O_Q     0ull                   // quantized_st  16,777,216
#define O_LOSS  16777216ull            // loss                  1
#define O_IDXO  16777217ull            // encoding_indices 32,768 (as float)
#define O_CS    16809985ull            // new_cs            8,192
#define O_EMAW  16818177ull            // new_ema_w     4,194,304
#define O_EMB   21012481ull            // new_embedding 4,194,304

// -------------------------------------------------------------------------
__global__ void vq_zero_kernel(float* __restrict__ p, size_t n) {
    size_t i = (size_t)blockIdx.x * blockDim.x + threadIdx.x;
    size_t stride = (size_t)gridDim.x * blockDim.x;
    for (; i < n; i += stride) p[i] = 0.0f;
}

// L2-normalize rows of [rows, 512]; emit f32 and f16 copies. One wave per row.
__global__ void vq_norm_kernel(const float* __restrict__ in,
                               float* __restrict__ of32,
                               _Float16* __restrict__ of16, int rows) {
    int wave = (int)((blockIdx.x * blockDim.x + threadIdx.x) >> 5);
    int lane = threadIdx.x & 31;
    if (wave >= rows) return;
    const float* p = in + (size_t)wave * DD + lane * 16;
    float v[16];
    float s = 0.0f;
#pragma unroll
    for (int i = 0; i < 16; ++i) { v[i] = p[i]; s += v[i] * v[i]; }
#pragma unroll
    for (int m = 1; m < 32; m <<= 1) s += __shfl_xor(s, m, 32);
    float r = 1.0f / fmaxf(sqrtf(s), 1e-12f);
    float*     q32 = of32 + (size_t)wave * DD + lane * 16;
    _Float16*  q16 = of16 + (size_t)wave * DD + lane * 16;
#pragma unroll
    for (int i = 0; i < 16; ++i) {
        float nv = v[i] * r;
        q32[i] = nv;
        q16[i] = (_Float16)nv;
    }
}

// -------------------------------------------------------------------------
// WMMA argmax-of-dot kernel: scores = flat @ emb^T, per-row argmax over K.
// Block: 8 waves x 16 rows = 128 rows. A resident in registers, B via LDS.
// Two independent WMMA chains (two 16-code sub-tiles) + 2-deep register
// double buffering of B fragments: loads for chunk ck+1 issue before the
// WMMAs of chunk ck, so each s_wait_dscnt has two WMMA executions of slack.
#define CTILE 32
#define BSTRIDE 520   // halves per LDS row (512 + 8 pad vs bank conflicts)

union Bfrag { v16h h; u32x4 u[2]; };

__device__ __forceinline__ void load_bfrag(Bfrag& b, const _Float16* p) {
    b.u[0] = *(const u32x4*)(p);
    b.u[1] = *(const u32x4*)(p + 8);
}

__global__ void __launch_bounds__(256)
vq_argmax_kernel(const _Float16* __restrict__ flat16,
                 const _Float16* __restrict__ emb16,
                 int* __restrict__ idx_out) {
    __shared__ _Float16 Bs[CTILE * BSTRIDE];
    const int lane = threadIdx.x & 31;
    const int wave = threadIdx.x >> 5;
    const int rowbase = blockIdx.x * 128 + wave * 16;
    const int hsel = (lane >> 4) & 1;

    // Resident A fragments: 16 chunks of K=32 over D=512.
    // A k order used here (contiguous): lanes 0-15 hold K 0..15, lanes 16-31 K 16..31 (matches B).
    const int arow = rowbase + (lane & 15);
    v16h afrag[16];
#pragma unroll
    for (int ck = 0; ck < 16; ++ck) {
        const _Float16* p = flat16 + (size_t)arow * DD + ck * 32 + hsel * 16;
        union { v16h h; u32x4 u[2]; } fr;
        fr.u[0] = *(const u32x4*)(p);
        fr.u[1] = *(const u32x4*)(p + 8);
        afrag[ck] = fr.h;
    }

    float best[8];
    int   bidx[8];
#pragma unroll
    for (int v = 0; v < 8; ++v) { best[v] = -1e30f; bidx[v] = 0; }

    for (int kt = 0; kt < KK; kt += CTILE) {
        __syncthreads();
        // Stage 32 codebook rows (32 x 512 halves) into LDS cooperatively.
        {
            const _Float16* src = emb16 + (size_t)kt * DD;
#pragma unroll
            for (int j = 0; j < 8; ++j) {
                int u = j * 256 + threadIdx.x;   // 2048 uint4 total
                int r = u >> 6;                  // 64 uint4 per 512-half row
                int c = u & 63;
                u32x4 d = *(const u32x4*)(src + (size_t)r * DD + c * 8);
                *(u32x4*)(&Bs[r * BSTRIDE + c * 8]) = d;
            }
        }
        if (kt + CTILE < KK) {
            // gfx1250 global_prefetch_b8 for the next tile (64 KB span)
            __builtin_prefetch(emb16 + (size_t)(kt + CTILE) * DD +
                               (size_t)threadIdx.x * 128, 0, 0);
        }
        __syncthreads();

        // B k order used here (contiguous, as A): lane = column, lane-half selects K 0-15 vs 16-31.
        const _Float16* brow0 = &Bs[(lane & 15) * BSTRIDE + hsel * 16];
        const _Float16* brow1 = brow0 + 16 * BSTRIDE;
        v8f acc0 = {};
        v8f acc1 = {};

        Bfrag b0[2], b1[2];
        load_bfrag(b0[0], brow0);
        load_bfrag(b1[0], brow1);
#pragma unroll
        for (int ck = 0; ck < 16; ++ck) {
            const int cur = ck & 1;
            const int nxt = cur ^ 1;
            if (ck < 15) {
                load_bfrag(b0[nxt], brow0 + (ck + 1) * 32);
                load_bfrag(b1[nxt], brow1 + (ck + 1) * 32);
            }
            acc0 = __builtin_amdgcn_wmma_f32_16x16x32_f16(
                false, afrag[ck], false, b0[cur].h, (short)0, acc0, false, false);
            acc1 = __builtin_amdgcn_wmma_f32_16x16x32_f16(
                false, afrag[ck], false, b1[cur].h, (short)0, acc1, false, false);
        }

        const int col0 = kt + (lane & 15);
        const int col1 = col0 + 16;
#pragma unroll
        for (int v = 0; v < 8; ++v) {
            float s0 = acc0[v];
            if (s0 > best[v]) { best[v] = s0; bidx[v] = col0; }
            float s1 = acc1[v];
            if (s1 > best[v]) { best[v] = s1; bidx[v] = col1; }
        }
    }

    // Per-row argmax reduction across the 16 lanes of each half-wave.
    // Tie-break: lower index wins (matches jnp.argmin first-occurrence).
#pragma unroll
    for (int v = 0; v < 8; ++v) {
        float b = best[v];
        int   bi = bidx[v];
#pragma unroll
        for (int m = 1; m < 16; m <<= 1) {
            float ob = __shfl_xor(b, m, 32);
            int   oi = __shfl_xor(bi, m, 32);
            if (ob > b || (ob == b && oi < bi)) { b = ob; bi = oi; }
        }
        if ((lane & 15) == 0) {
            int row = rowbase + v + 8 * hsel;
            idx_out[row] = bi;
        }
    }
}

// -------------------------------------------------------------------------
// Gather quantized rows, MSE loss partials, scatter counts + dw. Wave/row.
__global__ void vq_gather_kernel(const float* __restrict__ inputs,
                                 const float* __restrict__ embf32,
                                 const float* __restrict__ flatf32,
                                 const int* __restrict__ idx,
                                 float* __restrict__ out_q,
                                 float* __restrict__ out_idx,
                                 float* __restrict__ dw,
                                 float* __restrict__ counts,
                                 float* __restrict__ lossAcc) {
    int wave = (int)((blockIdx.x * blockDim.x + threadIdx.x) >> 5);
    int lane = threadIdx.x & 31;
    if (wave >= NN) return;
    int k = idx[wave];
    const float* e = embf32  + (size_t)k    * DD + lane * 16;
    const float* x = inputs  + (size_t)wave * DD + lane * 16;
    const float* f = flatf32 + (size_t)wave * DD + lane * 16;
    float*       q = out_q   + (size_t)wave * DD + lane * 16;
    float*       w = dw      + (size_t)k    * DD + lane * 16;
    float s = 0.0f;
#pragma unroll
    for (int i = 0; i < 16; ++i) {
        float ev = e[i];
        float d  = ev - x[i];
        s += d * d;
        q[i] = ev;                 // quantized_st == quantized numerically
        atomicAdd(&w[i], f[i]);    // dw = segment_sum(flat, idx)
    }
#pragma unroll
    for (int m = 1; m < 32; m <<= 1) s += __shfl_xor(s, m, 32);
    if (lane == 0) {
        atomicAdd(lossAcc, s);
        atomicAdd(&counts[k], 1.0f);
        out_idx[wave] = (float)k;
    }
}

// new_cs pre-normalization + global sum n
__global__ void vq_cs_kernel(const float* __restrict__ ema_cs,
                             const float* __restrict__ counts,
                             float* __restrict__ pre,
                             float* __restrict__ nAcc) {
    int i = blockIdx.x * blockDim.x + threadIdx.x;
    float p = 0.0f;
    if (i < KK) {
        p = ema_cs[i] * DECAYF + (1.0f - DECAYF) * counts[i];
        pre[i] = p;
    }
#pragma unroll
    for (int m = 1; m < 32; m <<= 1) p += __shfl_xor(p, m, 32);
    if ((threadIdx.x & 31) == 0) atomicAdd(nAcc, p);
}

// finalize new_cs / new_ema_w / new_embedding / loss
__global__ void vq_final_kernel(const float* __restrict__ ema_w,
                                const float* __restrict__ dw,
                                const float* __restrict__ pre,
                                const float* __restrict__ nAcc,
                                const float* __restrict__ lossAcc,
                                float* __restrict__ out_cs,
                                float* __restrict__ out_emaw,
                                float* __restrict__ out_emb,
                                float* __restrict__ out_loss) {
    size_t i = (size_t)blockIdx.x * blockDim.x + threadIdx.x;
    if (i >= (size_t)KK * DD) return;
    int k = (int)(i >> 9);
    float n = *nAcc;
    float cs = (pre[k] + EPSF) / (n + (float)KK * EPSF) * n;
    float nw = ema_w[i] * DECAYF + (1.0f - DECAYF) * dw[i];
    out_emaw[i] = nw;
    out_emb[i]  = nw / fmaxf(cs, EPSF);
    if ((i & 511) == 0) out_cs[k] = cs;
    if (i == 0) out_loss[0] = COMMITF * lossAcc[0] / (float)((size_t)NN * DD);
}

// -------------------------------------------------------------------------
extern "C" void kernel_launch(void* const* d_in, const int* in_sizes, int n_in,
                              void* d_out, int out_size, void* d_ws, size_t ws_size,
                              hipStream_t stream) {
    const float* inputs    = (const float*)d_in[0];
    const float* embedding = (const float*)d_in[1];
    const float* ema_cs    = (const float*)d_in[2];
    const float* ema_w     = (const float*)d_in[3];

    float* out = (float*)d_out;
    float* ws  = (float*)d_ws;

    float*    flatf32 = ws + O_FLATF32;
    float*    embf32  = ws + O_EMBF32;
    float*    dw      = ws + O_DW;
    float*    counts  = ws + O_COUNTS;
    float*    lossA   = ws + O_LOSSA;
    float*    nA      = ws + O_NA;
    float*    pre     = ws + O_PRE;
    int*      idxp    = (int*)(ws + O_IDX);
    _Float16* flat16  = (_Float16*)(ws + O_F16FLAT);
    _Float16* emb16   = (_Float16*)(ws + O_F16EMB);

    // zero dw + counts + lossAcc + nAcc (contiguous region)
    vq_zero_kernel<<<2048, 256, 0, stream>>>(dw, (size_t)KK * DD + KK + 2);

    vq_norm_kernel<<<NN / 8, 256, 0, stream>>>(inputs, flatf32, flat16, NN);
    vq_norm_kernel<<<KK / 8, 256, 0, stream>>>(embedding, embf32, emb16, KK);

    vq_argmax_kernel<<<NN / 128, 256, 0, stream>>>(flat16, emb16, idxp);

    vq_gather_kernel<<<NN / 8, 256, 0, stream>>>(inputs, embf32, flatf32, idxp,
                                                 out + O_Q, out + O_IDXO,
                                                 dw, counts, lossA);

    vq_cs_kernel<<<KK / 256, 256, 0, stream>>>(ema_cs, counts, pre, nA);

    vq_final_kernel<<<(KK * DD) / 256, 256, 0, stream>>>(
        ema_w, dw, pre, nA, lossA,
        out + O_CS, out + O_EMAW, out + O_EMB, out + O_LOSS);
}
